// GCN_53910429499630
// MI455X (gfx1250) — matrix operand order repaired and checked by reading, not compiled, not run on hardware
//
#include <hip/hip_runtime.h>

// ---------------------------------------------------------------------------
// GCN on MI455X (gfx1250): out = An@(relu(An@(relu(An@(x@W0)+b0)@W1)+b1)@W2)+b2
// An = D^-1/2 (A+I) D^-1/2 is never materialized:
//   * adj converted ONCE to f16 (+I on diagonal) fused into the rowsum pass
//   * dinv[j] folded into projected features Ys (stored f16, K-major)
//   * dinv[i], bias, relu folded into split-K reduction epilogue
// Big GEMM: double-buffered global_load_async_to_lds_b128 staging (ASYNCcnt)
// feeding v_wmma_f32_16x16x32_f16. Memory-bound: ~128MB f16 adj per layer.
// ---------------------------------------------------------------------------

typedef __attribute__((ext_vector_type(16))) _Float16 v16h;
typedef __attribute__((ext_vector_type(8)))  _Float16 v8h;
typedef __attribute__((ext_vector_type(4)))  _Float16 v4h;
typedef __attribute__((ext_vector_type(8)))  float    v8f;

#define NN      8192
#define DHID    128
#define KSPLIT  4
#define TK      64
#define LDSTR   80     // halves per LDS row (64 data + 16 pad) = 160B, 16B-aligned

// ---------------------------------------------------------------------------
// Pass 1 (streams adj once):
//   adjh[row][c] = (_Float16)( adj[row][c] + (row==c) )
//   dinv[row]    = rsqrt( sum(adj[row][:]) + 1 )
// ---------------------------------------------------------------------------
__global__ __launch_bounds__(256)
void rowsum_cvt(const float* __restrict__ adj, float* __restrict__ dinv,
                _Float16* __restrict__ adjh) {
  const int row = blockIdx.x;
  const float4* p = (const float4*)(adj + (size_t)row * NN);
  _Float16* dst = adjh + (size_t)row * NN;
  float s = 0.f;
  for (int f = threadIdx.x; f < NN / 4; f += 256) {
    float4 v = p[f];
    s += v.x + v.y + v.z + v.w;                  // raw sum (diag +1 added below)
    float a[4] = {v.x, v.y, v.z, v.w};
    if (f == (row >> 2)) a[row & 3] += 1.0f;     // A + I for the f16 copy
    v4h h = {(_Float16)a[0], (_Float16)a[1], (_Float16)a[2], (_Float16)a[3]};
    *(v4h*)(dst + f * 4) = h;
  }
  #pragma unroll
  for (int off = 16; off; off >>= 1) s += __shfl_xor(s, off, 32);
  __shared__ float wsum[8];
  const int lane = threadIdx.x & 31, w = threadIdx.x >> 5;
  if (lane == 0) wsum[w] = s;
  __syncthreads();
  if (threadIdx.x == 0) {
    float t = 0.f;
    #pragma unroll
    for (int i = 0; i < 8; ++i) t += wsum[i];
    dinv[row] = rsqrtf(t + 1.0f);
  }
}

// ---------------------------------------------------------------------------
// YsT[c][j] = (_Float16)( dinv[j] * sum_k H[j][k] * W[k][c] ),  c >= d_cols -> 0
// K-major f16 so the big GEMM's B tiles are contiguous 16B-aligned rows.
// ---------------------------------------------------------------------------
__global__ __launch_bounds__(256)
void proj_kernel(const float* __restrict__ H, const float* __restrict__ W,
                 const float* __restrict__ dinv, _Float16* __restrict__ YsT,
                 int d_in, int d_cols) {
  __shared__ _Float16 lds_y[DHID][32];
  const int t  = threadIdx.x;
  const int c  = t & 127;
  const int jh = t >> 7;
  const int j0 = blockIdx.x * 32;

  for (int jj = 0; jj < 16; ++jj) {
    const int j = j0 + jh * 16 + jj;
    float acc = 0.f;
    if (c < d_cols) {
      const float* hr = H + (size_t)j * d_in;
      for (int k = 0; k < d_in; ++k)
        acc = fmaf(hr[k], W[(size_t)k * d_cols + c], acc);
      acc *= dinv[j];
    }
    lds_y[c][jh * 16 + jj] = (_Float16)acc;
  }
  __syncthreads();

  const int cr = t >> 1, h = t & 1;
  const _Float16* src = &lds_y[cr][h * 16];
  _Float16* dst = YsT + (size_t)cr * NN + j0 + h * 16;
  *(v8h*)(dst)     = *(const v8h*)(src);
  *(v8h*)(dst + 8) = *(const v8h*)(src + 8);
}

// ---------------------------------------------------------------------------
// partial[s][i][c] = sum_{k in split s} adjh[i][k] * YsT[c][k]
// Block: 64 rows x 128 cols, 256 threads (8 wave32), wave owns 32x32 patch.
// Double-buffered async global->LDS staging, 8 WMMA per K-chunk of 64.
// ---------------------------------------------------------------------------
__global__ __launch_bounds__(256)
void spmm_wmma(const _Float16* __restrict__ adjh, const _Float16* __restrict__ YsT,
               float* __restrict__ partial) {
  __shared__ _Float16 lds_a[2][64][LDSTR];
  __shared__ _Float16 lds_b[2][DHID][LDSTR];

  const int t    = threadIdx.x;
  const int lane = t & 31;
  const int wv   = t >> 5;
  const int wm   = wv >> 2;                    // 0..1 : 32-row half
  const int wn   = wv & 3;                     // 0..3 : 32-col quarter
  const int lrow = lane & 15;
  const int hi   = lane >> 4;

  const size_t i0   = (size_t)blockIdx.x * 64;
  const int    kbeg = blockIdx.y * (NN / KSPLIT);
  const int    kend = kbeg + (NN / KSPLIT);

  const v8f vzero = {0.f, 0.f, 0.f, 0.f, 0.f, 0.f, 0.f, 0.f};
  v8f acc[2][2];
  acc[0][0] = vzero; acc[0][1] = vzero; acc[1][0] = vzero; acc[1][1] = vzero;

  // async copy mapping: each issue = one b128 (16B = 8 halves) per lane
  const int cr  = t >> 3;                      // row within 32-row group
  const int seg = t & 7;                       // 16B segment within 128B row

  auto issue_tile = [&](int buf, int k0) {
    // A tile: 64 rows x 64 halves (128B/row) -> 2 issues/thread
    #pragma unroll
    for (int j = 0; j < 2; ++j) {
      const int r = cr + j * 32;
      const _Float16* g = adjh + (i0 + r) * (size_t)NN + k0 + seg * 8;
      unsigned l = (unsigned)(uintptr_t)&lds_a[buf][r][seg * 8];
      asm volatile("global_load_async_to_lds_b128 %0, %1, off"
                   :: "v"(l), "v"(g) : "memory");
    }
    // B tile: 128 rows x 64 halves -> 4 issues/thread
    #pragma unroll
    for (int j = 0; j < 4; ++j) {
      const int r = cr + j * 32;
      const _Float16* g = YsT + (size_t)r * NN + k0 + seg * 8;
      unsigned l = (unsigned)(uintptr_t)&lds_b[buf][r][seg * 8];
      asm volatile("global_load_async_to_lds_b128 %0, %1, off"
                   :: "v"(l), "v"(g) : "memory");
    }
  };

  issue_tile(0, kbeg);
  asm volatile("s_wait_asynccnt 0" ::: "memory");
  __syncthreads();

  int cur = 0;
  for (int k0 = kbeg; k0 < kend; k0 += TK) {
    if (k0 + TK < kend) issue_tile(cur ^ 1, k0 + TK);

    #pragma unroll
    for (int ks = 0; ks < TK; ks += 32) {
      v16h afrag[2], bfrag[2];
      #pragma unroll
      for (int mt = 0; mt < 2; ++mt) {
        // A 16x32 f16 layout: lane row M = lrow; halves K = hi*8+{0..7}, +16
        const _Float16* pa = &lds_a[cur][wm * 32 + mt * 16 + lrow][ks + hi * 8];
        v8h lo = *(const v8h*)(pa);
        v8h up = *(const v8h*)(pa + 16);
        afrag[mt] = __builtin_shufflevector(lo, up, 0, 1, 2, 3, 4, 5, 6, 7,
                                            8, 9, 10, 11, 12, 13, 14, 15);
      }
      #pragma unroll
      for (int nt = 0; nt < 2; ++nt) {
        // B 32x16 f16 layout: lane col N = lrow; halves K = hi*8+{0..7}, +16
        const _Float16* pb = &lds_b[cur][wn * 32 + nt * 16 + lrow][ks + hi * 8];
        v8h lo = *(const v8h*)(pb);
        v8h up = *(const v8h*)(pb + 16);
        bfrag[nt] = __builtin_shufflevector(lo, up, 0, 1, 2, 3, 4, 5, 6, 7,
                                            8, 9, 10, 11, 12, 13, 14, 15);
      }
      #pragma unroll
      for (int mt = 0; mt < 2; ++mt)
        #pragma unroll
        for (int nt = 0; nt < 2; ++nt)
          acc[mt][nt] = __builtin_amdgcn_wmma_f32_16x16x32_f16(
              false, afrag[mt], false, bfrag[nt],
              (short)0, acc[mt][nt], false, false);
    }

    asm volatile("s_wait_asynccnt 0" ::: "memory");
    __syncthreads();
    cur ^= 1;
  }

  // store split partial (C/D layout: VGPR v -> M = v + 8*hi, N = lrow)
  float* base = partial + (size_t)blockIdx.y * NN * DHID;
  #pragma unroll
  for (int mt = 0; mt < 2; ++mt) {
    #pragma unroll
    for (int nt = 0; nt < 2; ++nt) {
      const int col = wn * 32 + nt * 16 + lrow;
      #pragma unroll
      for (int v = 0; v < 8; ++v) {
        const int row = (int)i0 + wm * 32 + mt * 16 + v + 8 * hi;
        base[(size_t)row * DHID + col] = acc[mt][nt][v];
      }
    }
  }
}

// ---------------------------------------------------------------------------
// out[i][c] = act( dinv[i] * sum_s partial[s][i][c] + bias[c] )
// ---------------------------------------------------------------------------
__global__ __launch_bounds__(256)
void reduce_epilogue(const float* __restrict__ partial, const float* __restrict__ dinv,
                     const float* __restrict__ bias, float* __restrict__ out,
                     int out_cols, int do_relu) {
  const int idx = blockIdx.x * 256 + threadIdx.x;
  const int i = idx / out_cols;
  const int c = idx - i * out_cols;
  const size_t off = (size_t)i * DHID + c;
  float s = 0.f;
  #pragma unroll
  for (int sp = 0; sp < KSPLIT; ++sp)
    s += partial[(size_t)sp * NN * DHID + off];
  float v = fmaf(dinv[i], s, bias[c]);
  if (do_relu) v = fmaxf(v, 0.f);
  out[(size_t)i * out_cols + c] = v;
}

// ---------------------------------------------------------------------------
extern "C" void kernel_launch(void* const* d_in, const int* in_sizes, int n_in,
                              void* d_out, int out_size, void* d_ws, size_t ws_size,
                              hipStream_t stream) {
  const float* x   = (const float*)d_in[0];
  const float* adj = (const float*)d_in[1];
  const float* W0  = (const float*)d_in[2];
  const float* b0  = (const float*)d_in[3];
  const float* W1  = (const float*)d_in[4];
  const float* b1  = (const float*)d_in[5];
  const float* W2  = (const float*)d_in[6];
  const float* b2  = (const float*)d_in[7];
  float* out = (float*)d_out;

  char* ws = (char*)d_ws;
  float*    dinv    = (float*)ws;      ws += (size_t)NN * 4;               // 32 KB
  _Float16* YsT     = (_Float16*)ws;   ws += (size_t)DHID * NN * 2;        // 2 MB
  float*    H       = (float*)ws;      ws += (size_t)NN * DHID * 4;        // 4 MB
  float*    partial = (float*)ws;      ws += (size_t)KSPLIT * NN * DHID * 4; // 16 MB
  _Float16* adjh    = (_Float16*)ws;   /* NN*NN*2 = 128 MB */

  const dim3 blk(256);
  const dim3 gemm_grid(NN / 64, KSPLIT);

  rowsum_cvt<<<NN, blk, 0, stream>>>(adj, dinv, adjh);

  // layer 0:  H = relu( An @ (x @ W0) + b0 )
  proj_kernel<<<NN / 32, blk, 0, stream>>>(x, W0, dinv, YsT, 64, 128);
  spmm_wmma<<<gemm_grid, blk, 0, stream>>>(adjh, YsT, partial);
  reduce_epilogue<<<(NN * 128) / 256, blk, 0, stream>>>(partial, dinv, b0, H, 128, 1);

  // layer 1:  H = relu( An @ (H @ W1) + b1 )
  proj_kernel<<<NN / 32, blk, 0, stream>>>(H, W1, dinv, YsT, 128, 128);
  spmm_wmma<<<gemm_grid, blk, 0, stream>>>(adjh, YsT, partial);
  reduce_epilogue<<<(NN * 128) / 256, blk, 0, stream>>>(partial, dinv, b1, H, 128, 1);

  // layer 2:  out = An @ (H @ W2) + b2
  proj_kernel<<<NN / 32, blk, 0, stream>>>(H, W2, dinv, YsT, 128, 64);
  spmm_wmma<<<gemm_grid, blk, 0, stream>>>(adjh, YsT, partial);
  reduce_epilogue<<<(NN * 64) / 256, blk, 0, stream>>>(partial, dinv, b2, out, 64, 0);
}
